// VQ_FNN_44057774522783
// MI455X (gfx1250) — matrix operand order repaired and checked by reading, not compiled, not run on hardware
//
#include <hip/hip_runtime.h>
#include <hip/hip_bf16.h>

// Problem dims (from reference)
#define OB_DIM     512
#define POLICY_DIM 256
#define N_RULES    16
#define ACTION_DIM 64
#define HIDDEN     64
#define BATCH      16384
#define XCOLS      (OB_DIM + POLICY_DIM)   // 768
#define NCOL       (N_RULES * ACTION_DIM)  // 1024 collapsed output columns
#define MROWS      32                      // batch rows per workgroup

typedef __attribute__((ext_vector_type(16))) __bf16 v16bf;
typedef __attribute__((ext_vector_type(8)))  float  v8f;

// -------- workspace layout --------
// [0, 1048576)       : WeffT  bf16 [NCOL][OB_DIM]   (transposed collapsed weights)
// [1048576, 1052672) : beff   f32  [NCOL]
// [1052672, 1069056) : invw   f32  [N_RULES*POLICY_DIM]
#define WS_WEFFT_OFF 0
#define WS_BEFF_OFF  (NCOL * OB_DIM * 2)
#define WS_INVW_OFF  (WS_BEFF_OFF + NCOL * 4)

// ---------------------------------------------------------------------------
// Prep: collapse the linear chain W1@W2@W3@W4 -> Weff (512x64 per rule),
// fold biases, reciprocal widths. One block per rule; tiny (~43 MFLOP total).
// ---------------------------------------------------------------------------
__global__ __launch_bounds__(256) void vqfnn_prep(
    const float* __restrict__ W1, const float* __restrict__ b1,
    const float* __restrict__ W2, const float* __restrict__ b2,
    const float* __restrict__ W3, const float* __restrict__ b3,
    const float* __restrict__ W4, const float* __restrict__ b4,
    const float* __restrict__ widths,
    __bf16* __restrict__ WeffT, float* __restrict__ beff,
    float* __restrict__ invw)
{
  __shared__ float M1[HIDDEN * HIDDEN];
  __shared__ float M2[HIDDEN * HIDDEN];
  __shared__ float cva[HIDDEN];
  __shared__ float cvb[HIDDEN];

  const int r   = blockIdx.x;
  const int tid = threadIdx.x;
  const float* w2 = W2 + r * HIDDEN * HIDDEN;
  const float* w3 = W3 + r * HIDDEN * HIDDEN;
  const float* w4 = W4 + r * HIDDEN * HIDDEN;

  // M1 = W2 @ W3
  for (int e = tid; e < HIDDEN * HIDDEN; e += 256) {
    int i = e >> 6, j = e & 63;
    float s = 0.f;
    for (int k = 0; k < HIDDEN; ++k) s = fmaf(w2[i * 64 + k], w3[k * 64 + j], s);
    M1[e] = s;
  }
  __syncthreads();
  // M2 = M1 @ W4
  for (int e = tid; e < HIDDEN * HIDDEN; e += 256) {
    int i = e >> 6, j = e & 63;
    float s = 0.f;
    for (int k = 0; k < HIDDEN; ++k) s = fmaf(M1[i * 64 + k], w4[k * 64 + j], s);
    M2[e] = s;
  }
  __syncthreads();
  // WeffT[r*64 + a][o] = sum_h W1[r][o][h] * M2[h][a]   (store transposed, bf16)
  for (int e = tid; e < OB_DIM * ACTION_DIM; e += 256) {
    int o = e >> 6, a = e & 63;
    const float* w1row = W1 + ((size_t)r * OB_DIM + o) * HIDDEN;
    float s = 0.f;
    for (int h = 0; h < HIDDEN; ++h) s = fmaf(w1row[h], M2[h * 64 + a], s);
    WeffT[(size_t)(r * ACTION_DIM + a) * OB_DIM + o] = (__bf16)s;
  }
  // bias chain: beff = ((b1@W2 + b2)@W3 + b3)@W4 + b4
  if (tid < HIDDEN) cva[tid] = b1[r * HIDDEN + tid];
  __syncthreads();
  if (tid < HIDDEN) {
    float s = b2[r * HIDDEN + tid];
    for (int h = 0; h < HIDDEN; ++h) s = fmaf(cva[h], w2[h * 64 + tid], s);
    cvb[tid] = s;
  }
  __syncthreads();
  if (tid < HIDDEN) {
    float s = b3[r * HIDDEN + tid];
    for (int h = 0; h < HIDDEN; ++h) s = fmaf(cvb[h], w3[h * 64 + tid], s);
    cva[tid] = s;
  }
  __syncthreads();
  if (tid < HIDDEN) {
    float s = b4[r * HIDDEN + tid];
    for (int h = 0; h < HIDDEN; ++h) s = fmaf(cva[h], w4[h * 64 + tid], s);
    beff[r * HIDDEN + tid] = s;
  }
  // reciprocal widths
  invw[r * POLICY_DIM + tid] = 1.0f / widths[r * POLICY_DIM + tid];
}

// ---------------------------------------------------------------------------
// Main fused kernel: one block per 32 batch rows (2 M-tiles).
//   - stage xo in LDS as bf16 WMMA-A rows (padded, b128-aligned)
//   - rule scores z read xp straight from global (broadcast, L2-resident)
//   - GEMM 32x1024: wave w owns cols [128w,128w+128) = rules 2w,2w+1;
//     each B fragment feeds TWO v_wmma_f32_16x16x32_bf16 (both row tiles);
//     B fragments are explicitly ping-pong double-buffered so the next
//     tile's global_load_b128 pair is in flight during the current WMMAs
//   - register-resident action softmax per (row,rule) via 16-lane shfl_xor
//     butterflies; both rules' fs-weighted contributions merged before the
//     ds_add_f32 accumulation into fnn[32][64]
//   - final softmax over 64 actions, coalesced store
// ---------------------------------------------------------------------------
__global__ __launch_bounds__(256) void vqfnn_main(
    const float* __restrict__ X, const float* __restrict__ centers,
    const __bf16* __restrict__ WeffT, const float* __restrict__ beff,
    const float* __restrict__ invw, float* __restrict__ out)
{
  __shared__ __bf16 Abuf[MROWS][OB_DIM + 8];   // 32 x 520 bf16 (33280 B)
  __shared__ float  zmat[MROWS][16];           // 2 KB
  __shared__ float  fsmat[MROWS][16];          // 2 KB
  __shared__ float  fnn[MROWS][ACTION_DIM];    // 8 KB
  __shared__ float  rmax[MROWS], rsinv[MROWS];

  const int tid = threadIdx.x;
  const int b0  = blockIdx.x * MROWS;

  // ---- stage xo tile: 32 rows x 128 float4 -> bf16 ----
  {
    const float4* X4 = (const float4*)X + (size_t)b0 * (XCOLS / 4);
    for (int i = tid; i < MROWS * (OB_DIM / 4); i += 256) {
      int row = i >> 7, c4 = i & 127;
      float4 v = X4[(size_t)row * (XCOLS / 4) + c4];
      union { __bf16 h[4]; uint2 u; } p;
      p.h[0] = (__bf16)v.x; p.h[1] = (__bf16)v.y;
      p.h[2] = (__bf16)v.z; p.h[3] = (__bf16)v.w;
      *(uint2*)&Abuf[row][c4 * 4] = p.u;
    }
  }

  // ---- rule scores: 512 (row,rule) pairs, xp read from global ----
  for (int p = tid; p < MROWS * N_RULES; p += 256) {
    const int row = p >> 4, rule = p & 15;
    const float* xp  = X + (size_t)(b0 + row) * XCOLS + OB_DIM;
    const float* cen = centers + rule * POLICY_DIM;
    const float* iw  = invw    + rule * POLICY_DIM;
    float s = 0.f;
    for (int k = 0; k < POLICY_DIM; ++k) {
      float d = (xp[k] - cen[k]) * iw[k];
      s = fmaf(d, d, s);
    }
    zmat[row][rule] = s * (-0.5f / (float)POLICY_DIM);
  }
  __syncthreads();

  // ---- rule softmax (32 threads, one per row) + zero fnn ----
  if (tid < MROWS) {
    float mx = -1e30f;
    #pragma unroll
    for (int r = 0; r < 16; ++r) mx = fmaxf(mx, zmat[tid][r]);
    float e[16], s = 0.f;
    #pragma unroll
    for (int r = 0; r < 16; ++r) { e[r] = __expf(zmat[tid][r] - mx); s += e[r]; }
    float inv = 1.f / s;
    #pragma unroll
    for (int r = 0; r < 16; ++r) fsmat[tid][r] = e[r] * inv;
  }
  for (int i = tid; i < MROWS * ACTION_DIM; i += 256) ((float*)fnn)[i] = 0.f;
  __syncthreads();

  // ---- WMMA GEMM: 2 row-tiles x 8 col-tiles per wave over K=512 ----
  const int wave = tid >> 5, lane = tid & 31;
  const int m  = lane & 15;   // A row / B col within tile
  const int hi = lane >> 4;   // lane half (K-swizzle selector)

  union Frag { v16bf v; uint4 u[2]; };
  v8f acc[2][8] = {};

  // B fragment (16-bit B 32x16): lane n=m, half hi holds K=16hi..16hi+15,
  // contiguous in the transposed weight storage.
  const uint4* Bbase = (const uint4*)(WeffT + (size_t)(wave * 128 + m) * OB_DIM
                                            + hi * 16);
  // tile t, chunk kc -> uint4 index: t*(16*OB_DIM/8) + kc*4
  Frag bb[2];
  bb[0].u[0] = Bbase[0];
  bb[0].u[1] = Bbase[1];

  for (int kc = 0; kc < 16; ++kc) {
    // A fragments (16-bit A 16x32): lane(m,hi) holds K = 16hi..16hi+15 (matches B)
    Frag a0, a1;
    a0.u[0] = *(const uint4*)&Abuf[m     ][kc * 32 + 16 * hi];
    a0.u[1] = *(const uint4*)&Abuf[m     ][kc * 32 + 16 * hi + 8];
    a1.u[0] = *(const uint4*)&Abuf[m + 16][kc * 32 + 16 * hi];
    a1.u[1] = *(const uint4*)&Abuf[m + 16][kc * 32 + 16 * hi + 8];
    #pragma unroll
    for (int t = 0; t < 8; ++t) {
      // Prefetch next tile's B fragment (tile t+1, wrapping to kc+1 at t==7;
      // the final iteration over-reads 32B into beff -- valid ws memory).
      {
        const int pf  = kc * 8 + t + 1;           // linear (kc,t) prefetch index
        const int pkc = pf >> 3, pt = pf & 7;
        const size_t bi = (size_t)pt * (16 * OB_DIM / 8) + (size_t)pkc * 4;
        bb[(t + 1) & 1].u[0] = Bbase[bi];
        bb[(t + 1) & 1].u[1] = Bbase[bi + 1];
      }
      acc[0][t] = __builtin_amdgcn_wmma_f32_16x16x32_bf16(
          false, a0.v, false, bb[t & 1].v, (short)0, acc[0][t], false, false);
      acc[1][t] = __builtin_amdgcn_wmma_f32_16x16x32_bf16(
          false, a1.v, false, bb[t & 1].v, (short)0, acc[1][t], false, false);
    }
  }

  // ---- bias add (bias depends only on column) ----
  #pragma unroll
  for (int t = 0; t < 8; ++t) {
    float bv = beff[wave * 128 + t * 16 + m];
    #pragma unroll
    for (int j = 0; j < 8; ++j) { acc[0][t][j] += bv; acc[1][t][j] += bv; }
  }

  // ---- per-(row,rule) action softmax, fs-weighted accumulate ----
  // D layout: vgpr j, lanes 0-15 -> M=j, lanes 16-31 -> M=j+8; N = lane&15.
  // Rule 2w covers tiles 0..3 (a = q*16+n); rule 2w+1 covers tiles 4..7.
  const int r0 = wave * 2, r1 = r0 + 1;
  #pragma unroll
  for (int rt = 0; rt < 2; ++rt) {
    #pragma unroll
    for (int j = 0; j < 8; ++j) {
      const int row = rt * 16 + j + 8 * hi;
      float y0 = acc[rt][0][j], y1 = acc[rt][1][j];
      float y2 = acc[rt][2][j], y3 = acc[rt][3][j];
      float u0 = acc[rt][4][j], u1 = acc[rt][5][j];
      float u2 = acc[rt][6][j], u3 = acc[rt][7][j];
      float mA = fmaxf(fmaxf(y0, y1), fmaxf(y2, y3));
      float mB = fmaxf(fmaxf(u0, u1), fmaxf(u2, u3));
      #pragma unroll
      for (int mask = 1; mask < 16; mask <<= 1) {   // stays inside 16-lane half
        mA = fmaxf(mA, __shfl_xor(mA, mask, 32));
        mB = fmaxf(mB, __shfl_xor(mB, mask, 32));
      }
      float eA0 = __expf(y0 - mA), eA1 = __expf(y1 - mA);
      float eA2 = __expf(y2 - mA), eA3 = __expf(y3 - mA);
      float eB0 = __expf(u0 - mB), eB1 = __expf(u1 - mB);
      float eB2 = __expf(u2 - mB), eB3 = __expf(u3 - mB);
      float sA = eA0 + eA1 + eA2 + eA3;
      float sB = eB0 + eB1 + eB2 + eB3;
      #pragma unroll
      for (int mask = 1; mask < 16; mask <<= 1) {
        sA += __shfl_xor(sA, mask, 32);
        sB += __shfl_xor(sB, mask, 32);
      }
      float wA = fsmat[row][r0] / sA;
      float wB = fsmat[row][r1] / sB;
      atomicAdd(&fnn[row][ 0 + m], eA0 * wA + eB0 * wB);
      atomicAdd(&fnn[row][16 + m], eA1 * wA + eB1 * wB);
      atomicAdd(&fnn[row][32 + m], eA2 * wA + eB2 * wB);
      atomicAdd(&fnn[row][48 + m], eA3 * wA + eB3 * wB);
    }
  }
  __syncthreads();

  // ---- final softmax over 64 actions ----
  if (tid < MROWS) {
    float mx = -1e30f;
    #pragma unroll
    for (int a = 0; a < ACTION_DIM; ++a) mx = fmaxf(mx, fnn[tid][a]);
    float s = 0.f;
    #pragma unroll
    for (int a = 0; a < ACTION_DIM; ++a) s += __expf(fnn[tid][a] - mx);
    rmax[tid] = mx; rsinv[tid] = 1.f / s;
  }
  __syncthreads();
  for (int i = tid; i < MROWS * ACTION_DIM; i += 256) {
    int row = i >> 6, a = i & 63;
    out[(size_t)(b0 + row) * ACTION_DIM + a] =
        __expf(fnn[row][a] - rmax[row]) * rsinv[row];
  }
}

// ---------------------------------------------------------------------------
extern "C" void kernel_launch(void* const* d_in, const int* in_sizes, int n_in,
                              void* d_out, int out_size, void* d_ws, size_t ws_size,
                              hipStream_t stream) {
  // setup_inputs order: X, centers, widths, W1, b1, W2, b2, W3, b3, W4, b4
  const float* X       = (const float*)d_in[0];
  const float* centers = (const float*)d_in[1];
  const float* widths  = (const float*)d_in[2];
  const float* W1 = (const float*)d_in[3];
  const float* b1 = (const float*)d_in[4];
  const float* W2 = (const float*)d_in[5];
  const float* b2 = (const float*)d_in[6];
  const float* W3 = (const float*)d_in[7];
  const float* b3 = (const float*)d_in[8];
  const float* W4 = (const float*)d_in[9];
  const float* b4 = (const float*)d_in[10];
  float* out = (float*)d_out;

  char* ws = (char*)d_ws;
  __bf16* WeffT = (__bf16*)(ws + WS_WEFFT_OFF);
  float*  beff  = (float*)(ws + WS_BEFF_OFF);
  float*  invw  = (float*)(ws + WS_INVW_OFF);

  vqfnn_prep<<<N_RULES, 256, 0, stream>>>(W1, b1, W2, b2, W3, b3, W4, b4,
                                          widths, WeffT, beff, invw);
  vqfnn_main<<<BATCH / MROWS, 256, 0, stream>>>(X, centers, WeffT, beff, invw, out);
}
